// SelfAttentive_82867099009604
// MI455X (gfx1250) — matrix operand order repaired and checked by reading, not compiled, not run on hardware
//
#include <hip/hip_runtime.h>

// ---------------- types ----------------
typedef __bf16        bf16x16 __attribute__((ext_vector_type(16)));
typedef float         f32x8   __attribute__((ext_vector_type(8)));
typedef unsigned int  u32x4   __attribute__((ext_vector_type(4)));

// ---------------- problem constants ----------------
constexpr int kV    = 50000;
constexpr int kEMB  = 300;
constexpr int kEMBP = 320;        // EMB padded to multiple of 32
constexpr int kB    = 64;
constexpr int kS    = 512;
constexpr int kNH   = 256;
constexpr int kDA   = 128;
constexpr int kR    = 16;
constexpr int kOUT  = 2;
constexpr int kKC   = kEMBP + kNH;   // 576 fused K for [x|h] GEMM
constexpr int kH2   = 2 * kNH;       // 512

// ---------------- workspace layout (bytes, all 256B aligned) ----------------
constexpr size_t SZ_WC  = (size_t)4 * kNH * kKC * 2;          // 1024x576 bf16
constexpr size_t OFF_WCF = 0;
constexpr size_t OFF_WCB = OFF_WCF + SZ_WC;
constexpr size_t OFF_S1  = OFF_WCB + SZ_WC;
constexpr size_t OFF_S2  = OFF_S1 + (size_t)kDA * kH2 * 2;
constexpr size_t OFF_MW  = OFF_S2 + (size_t)kR * kDA * 2;
constexpr size_t OFF_XF  = OFF_MW + (size_t)kNH * (kR * kH2) * 2;
constexpr size_t OFF_XR  = OFF_XF + (size_t)kS * kB * kEMBP * 2;
constexpr size_t OFF_H   = OFF_XR + (size_t)kS * kB * kEMBP * 2;
constexpr size_t OFF_LG  = OFF_H  + (size_t)kB * kS * kH2 * 2;
constexpr size_t OFF_BM  = OFF_LG + (size_t)kB * kR * kS * 4;

// ---------------- helpers ----------------
__device__ __forceinline__ unsigned short f2bf(float f) {
  union { float f; unsigned u; } v; v.f = f;
  unsigned u = v.u;
  return (unsigned short)((u + 0x7FFFu + ((u >> 16) & 1u)) >> 16);
}
__device__ __forceinline__ float bf2f(unsigned short h) {
  union { unsigned u; float f; } v; v.u = ((unsigned)h) << 16; return v.f;
}
__device__ __forceinline__ f32x8 fzero() {
  f32x8 z;
#pragma unroll
  for (int i = 0; i < 8; ++i) z[i] = 0.f;
  return z;
}
// Opaque zero: re-evaluated by the asm every loop iteration, so loads indexed
// by it cannot be hoisted out of the loop (blocks LICM + scratch-caching of
// the weight stream) while keeping pointer provenance -> global_load, not flat.
__device__ __forceinline__ int opaque_zero() {
  int z = 0;
  asm volatile("" : "+v"(z));
  return z;
}
// A-operand tile (16x32 bf16, row-major memory, row stride = `stride` elems).
// ISA layout: lane<16 holds row (lane) K[0..7]+K[16..23]; lane>=16 K[8..15]+K[24..31].
__device__ __forceinline__ bf16x16 ldA(const unsigned short* base, int stride, int lane) {
  const unsigned short* p = base + (size_t)(lane & 15) * stride + ((lane >> 4) << 3);
  union { bf16x16 v; u32x4 q[2]; } t;
  t.q[0] = *reinterpret_cast<const u32x4*>(p);
  t.q[1] = *reinterpret_cast<const u32x4*>(p + 16);
  return t.v;
}
// B-operand tile (32x16). For C = X * W^T with W row-major (N x K, stride K),
// column n of B is row n of W: lane<16 holds K[0..7]+K[16..23] of col (lane),
// lane>=16 holds K[8..15]+K[24..31] of col (lane-16).
__device__ __forceinline__ bf16x16 ldB(const unsigned short* base, int stride, int lane) {
  const unsigned short* p = base + (size_t)(lane & 15) * stride + ((lane >> 4) << 3);
  union { bf16x16 v; u32x4 q[2]; } t;
  t.q[0] = *reinterpret_cast<const u32x4*>(p);
  t.q[1] = *reinterpret_cast<const u32x4*>(p + 16);
  return t.v;
}
__device__ __forceinline__ f32x8 wmma_bf16(bf16x16 a, bf16x16 b, f32x8 c) {
  return __builtin_amdgcn_wmma_f32_16x16x32_bf16(false, a, false, b, (short)0, c, false, false);
}
__device__ __forceinline__ float sigm(float x) { return 1.0f / (1.0f + __expf(-x)); }

// ---------------- weight prep ----------------
// Fuse [Wih(pad 300->320) | Whh] into (1024 x 576) bf16 with gate-interleaved
// rows: out row n' = 4*j + g  <- source row g*256 + j  (g in {i,f,g,o}).
__global__ void pack_wcomb_kernel(const float* __restrict__ Wih,
                                  const float* __restrict__ Whh,
                                  unsigned short* __restrict__ dst) {
  int np = blockIdx.x;                 // 0..1023
  int j = np >> 2, g = np & 3;
  int src = g * kNH + j;
  int k = threadIdx.x;                 // 0..575
  float v;
  if (k < kEMB)        v = Wih[(size_t)src * kEMB + k];
  else if (k < kEMBP)  v = 0.f;
  else                 v = Whh[(size_t)src * kNH + (k - kEMBP)];
  dst[(size_t)np * kKC + k] = f2bf(v);
}

__global__ void cvt_bf16_kernel(const float* __restrict__ s,
                                unsigned short* __restrict__ d, int n) {
  int i = blockIdx.x * blockDim.x + threadIdx.x;
  if (i < n) d[i] = f2bf(s[i]);
}

// ---------------- embedding gather (fwd + per-example reversed), time-major ----------------
__global__ __launch_bounds__(320, 1)
void embed_kernel(const int* __restrict__ x, const int* __restrict__ seq,
                  const float* __restrict__ emb,
                  unsigned short* __restrict__ XF, unsigned short* __restrict__ XR) {
  int t = blockIdx.x, b = blockIdx.y, e = threadIdx.x;
  int tokf = x[b * kS + t];
  int len  = seq[b];
  int tr   = (t < len) ? (len - 1 - t) : t;
  int tokr = x[b * kS + tr];
  size_t row = ((size_t)t * kB + b) * kEMBP;
  XF[row + e] = (e < kEMB) ? f2bf(emb[(size_t)tokf * kEMB + e]) : (unsigned short)0;
  XR[row + e] = (e < kEMB) ? f2bf(emb[(size_t)tokr * kEMB + e]) : (unsigned short)0;
}

// ---------------- persistent bidirectional LSTM ----------------
// grid.x = 2 (direction). One workgroup (1024 thr = 32 waves) runs the whole
// recurrence for its direction. Per step: Z(64x1024) = [x_t|h] @ Wcomb^T via
// WMMA; gate quads resolved with 4-lane shuffles.
// c lives in LDS (owned per-wave); 8 n-tiles per wave processed in 2 chunks
// of 4 (only 4 accumulators live); h double-buffered in LDS -> one barrier
// per step. Weight stream re-read from L2 each step via opaque-offset
// laundering (global_load, no scratch caching); B tiles issued in pairs to
// widen each wait-group.
__global__ __launch_bounds__(1024, 1)
void lstm_kernel(const unsigned short* __restrict__ XF,
                 const unsigned short* __restrict__ XR,
                 const unsigned short* __restrict__ WF,
                 const unsigned short* __restrict__ WB,
                 const int* __restrict__ seq,
                 unsigned short* __restrict__ H) {
  const int dir = blockIdx.x;
  const unsigned short* X = dir ? XR : XF;
  const unsigned short* W = dir ? WB : WF;
  const int dOff = dir ? kNH : 0;

  __shared__ unsigned short hbuf[2][kB * kNH];   // 2 x 32 KB double-buffered h
  __shared__ float          c_lds[kB * kNH];     // 64 KB cell state
  __shared__ int            seq_lds[kB];

  const int tid = threadIdx.x, lane = tid & 31, wave = tid >> 5;
  const int mt = wave >> 3, m0 = mt << 4;        // batch tile
  const int ng = wave & 7;                       // n-group: 8 n-tiles each

  for (int i = tid; i < kB * kNH; i += 1024) {
    hbuf[0][i] = 0;
    c_lds[i]   = 0.f;
  }
  if (tid < kB) seq_lds[tid] = seq[tid];
  __syncthreads();

  const int brow_base = m0 + ((lane >> 4) << 3);
  const int base = lane & ~3;
  int cur = 0;

  for (int t = 0; t < kS; ++t) {
    // fresh opaque zero each step: blocks hoisting of the weight tiles out of
    // the t-loop while keeping global address-space provenance
    const int zz = opaque_zero();
    const unsigned short* Wl   = W + zz;
    const unsigned short* xrow = X + ((size_t)t * kB + m0) * kEMBP + zz;
    const unsigned short* hold = &hbuf[cur][0];
    unsigned short*       hnew = &hbuf[cur ^ 1][0];

    if (t + 1 < kS)  // warm next step's activation rows into cache
      __builtin_prefetch(xrow + (size_t)kB * kEMBP + (lane << 4), 0, 1);

#pragma unroll
    for (int chunk = 0; chunk < 2; ++chunk) {
      const unsigned short* wbase =
          Wl + (size_t)(((ng << 3) + (chunk << 2)) << 4) * kKC;
      f32x8 acc[4];
#pragma unroll
      for (int q = 0; q < 4; ++q) acc[q] = fzero();

      // ---- K over x_t (320) ----
      for (int kk = 0; kk < kEMBP; kk += 32) {
        bf16x16 a  = ldA(xrow + kk, kEMBP, lane);
        bf16x16 b0 = ldB(wbase + (size_t)(0 * 16) * kKC + kk, kKC, lane);
        bf16x16 b1 = ldB(wbase + (size_t)(1 * 16) * kKC + kk, kKC, lane);
        acc[0] = wmma_bf16(a, b0, acc[0]);
        acc[1] = wmma_bf16(a, b1, acc[1]);
        bf16x16 b2 = ldB(wbase + (size_t)(2 * 16) * kKC + kk, kKC, lane);
        bf16x16 b3 = ldB(wbase + (size_t)(3 * 16) * kKC + kk, kKC, lane);
        acc[2] = wmma_bf16(a, b2, acc[2]);
        acc[3] = wmma_bf16(a, b3, acc[3]);
      }
      // ---- K over h (256), A from LDS (previous step's buffer) ----
      for (int kk = 0; kk < kNH; kk += 32) {
        bf16x16 a  = ldA(hold + m0 * kNH + kk, kNH, lane);
        bf16x16 b0 = ldB(wbase + (size_t)(0 * 16) * kKC + kEMBP + kk, kKC, lane);
        bf16x16 b1 = ldB(wbase + (size_t)(1 * 16) * kKC + kEMBP + kk, kKC, lane);
        acc[0] = wmma_bf16(a, b0, acc[0]);
        acc[1] = wmma_bf16(a, b1, acc[1]);
        bf16x16 b2 = ldB(wbase + (size_t)(2 * 16) * kKC + kEMBP + kk, kKC, lane);
        bf16x16 b3 = ldB(wbase + (size_t)(3 * 16) * kKC + kEMBP + kk, kKC, lane);
        acc[2] = wmma_bf16(a, b2, acc[2]);
        acc[3] = wmma_bf16(a, b3, acc[3]);
      }

      // ---- gates: columns are 4*j+g, quads live in 4 adjacent lanes ----
#pragma unroll
      for (int q = 0; q < 4; ++q) {
        const int n0 = ((ng << 3) + (chunk << 2) + q) << 4;
        const int jcol = (n0 + ((lane & 15) & ~3)) >> 2;
#pragma unroll
        for (int r = 0; r < 8; ++r) {
          float z  = acc[q][r];
          float zi = __shfl(z, base + 0, 32);
          float zf = __shfl(z, base + 1, 32);
          float zg = __shfl(z, base + 2, 32);
          float zo = __shfl(z, base + 3, 32);
          int brow = brow_base + r;
          float c_old = c_lds[brow * kNH + jcol];   // broadcast read (owned by this wave)
          float c_new = sigm(zf) * c_old + sigm(zi) * tanhf(zg);
          float h     = sigm(zo) * tanhf(c_new);
          if ((lane & 3) == 0) {
            c_lds[brow * kNH + jcol] = c_new;
            unsigned short hb = f2bf(h);
            hnew[brow * kNH + jcol] = hb;
            int tpos = t;
            if (dir) { int len = seq_lds[brow]; tpos = (t < len) ? (len - 1 - t) : t; }
            H[(size_t)brow * kS * kH2 + (size_t)tpos * kH2 + dOff + jcol] = hb;
          }
        }
      }
    }
    __syncthreads();   // h_new complete before it becomes h_old of step t+1
    cur ^= 1;
  }
}

// ---------------- attention scores: tanh(H @ S1^T) @ S2^T, masked ----------------
__global__ __launch_bounds__(256, 1)
void attn_kernel(const unsigned short* __restrict__ H,
                 const unsigned short* __restrict__ S1,
                 const unsigned short* __restrict__ S2,
                 const int* __restrict__ seq,
                 float* __restrict__ logits) {
  __shared__ unsigned short tl[kB * kDA];        // tanh(H S1^T), bf16
  const int tid = threadIdx.x, lane = tid & 31, wave = tid >> 5;
  const size_t rowbase = (size_t)blockIdx.x * 64;   // flattened (b*S+t) rows

  { // stage 1: (64 x 512) x (512 x 128)
    const int mt = wave >> 1, m0 = mt << 4, nh = wave & 1;
    f32x8 acc[4];
#pragma unroll
    for (int q = 0; q < 4; ++q) acc[q] = fzero();
    for (int kk = 0; kk < kH2; kk += 32) {
      bf16x16 a = ldA(H + (rowbase + m0) * kH2 + kk, kH2, lane);
#pragma unroll
      for (int q = 0; q < 4; ++q) {
        int n0 = ((nh << 2) + q) << 4;
        bf16x16 b = ldB(S1 + (size_t)n0 * kH2 + kk, kH2, lane);
        acc[q] = wmma_bf16(a, b, acc[q]);
      }
    }
    int rofs = (lane >> 4) << 3;
#pragma unroll
    for (int q = 0; q < 4; ++q) {
      int col = ((((nh << 2) + q) << 4)) + (lane & 15);
#pragma unroll
      for (int r = 0; r < 8; ++r)
        tl[(m0 + rofs + r) * kDA + col] = f2bf(tanhf(acc[q][r]));
    }
  }
  __syncthreads();

  if (wave < 4) { // stage 2: (64 x 128) x (128 x 16)
    const int m0 = wave << 4;
    f32x8 acc = fzero();
    for (int kk = 0; kk < kDA; kk += 32) {
      bf16x16 a = ldA(tl + m0 * kDA + kk, kDA, lane);
      bf16x16 b = ldB(S2 + kk, kDA, lane);
      acc = wmma_bf16(a, b, acc);
    }
    int rcol = lane & 15, rofs = (lane >> 4) << 3;
#pragma unroll
    for (int r = 0; r < 8; ++r) {
      int row = (int)rowbase + m0 + rofs + r;
      int b = row >> 9, t = row & (kS - 1);
      float v = acc[r];
      if (t >= seq[b]) v = -1.0e9f;
      logits[((size_t)b * kR + rcol) * kS + t] = v;
    }
  }
}

// ---------------- softmax over S per (b,r) ----------------
__global__ __launch_bounds__(512, 1)
void softmax_kernel(const float* __restrict__ logits, float* __restrict__ outA) {
  __shared__ float red[512];
  int i = threadIdx.x;
  size_t base = (size_t)blockIdx.x * kS;
  float v = logits[base + i];
  red[i] = v; __syncthreads();
  for (int s = 256; s > 0; s >>= 1) { if (i < s) red[i] = fmaxf(red[i], red[i + s]); __syncthreads(); }
  float mx = red[0]; __syncthreads();
  float e = __expf(v - mx);
  red[i] = e; __syncthreads();
  for (int s = 256; s > 0; s >>= 1) { if (i < s) red[i] += red[i + s]; __syncthreads(); }
  outA[base + i] = e / red[0];
}

// ---------------- M = A @ H, flattened straight into bf16 BM ----------------
__global__ __launch_bounds__(512, 1)
void bm_kernel(const float* __restrict__ A, const unsigned short* __restrict__ H,
               unsigned short* __restrict__ BM) {
  int d = threadIdx.x;
  int b = blockIdx.x >> 4;
  const float* arow = A + (size_t)blockIdx.x * kS;
  const unsigned short* hb = H + (size_t)b * kS * kH2;
  float acc = 0.f;
  for (int s = 0; s < kS; ++s) acc += arow[s] * bf2f(hb[(size_t)s * kH2 + d]);
  BM[(size_t)blockIdx.x * kH2 + d] = f2bf(acc);   // == BM[b][r*512+d]
}

// ---------------- penalty ----------------
__global__ void zero_penal_kernel(float* p) { if (threadIdx.x == 0) *p = 0.f; }

__global__ __launch_bounds__(256, 1)
void aat_kernel(const float* __restrict__ A, float* __restrict__ penal) {
  __shared__ float red[256];
  int b = blockIdx.x, r = threadIdx.x >> 4, k = threadIdx.x & 15;
  const float* Ab = A + (size_t)b * kR * kS;
  float dot = 0.f;
  for (int s = 0; s < kS; ++s) dot += Ab[r * kS + s] * Ab[k * kS + s];
  float diff = dot - (r == k ? 1.f : 0.f);
  red[threadIdx.x] = diff * diff;
  __syncthreads();
  for (int s = 128; s > 0; s >>= 1) { if (threadIdx.x < s) red[threadIdx.x] += red[threadIdx.x + s]; __syncthreads(); }
  if (threadIdx.x == 0) atomicAdd(penal, red[0] * (1.0f / kB));
}

// ---------------- MLP (WMMA) + decoder ----------------
__global__ __launch_bounds__(512, 1)
void mlp_dec_kernel(const unsigned short* __restrict__ BM,
                    const unsigned short* __restrict__ Wm,
                    const float* __restrict__ bias,
                    const float* __restrict__ dw, const float* __restrict__ db,
                    float* __restrict__ out) {
  __shared__ unsigned short hid[kB * kNH];   // relu output, bf16
  const int tid = threadIdx.x, lane = tid & 31, wave = tid >> 5;
  constexpr int KK = kR * kH2;               // 8192
  {
    const int mt = wave >> 2, m0 = mt << 4, g4 = wave & 3;
    f32x8 acc[4];
#pragma unroll
    for (int q = 0; q < 4; ++q) acc[q] = fzero();
    for (int kk = 0; kk < KK; kk += 32) {
      bf16x16 a = ldA(BM + (size_t)m0 * KK + kk, KK, lane);
#pragma unroll
      for (int q = 0; q < 4; ++q) {
        int n0 = ((g4 << 2) + q) << 4;
        bf16x16 b = ldB(Wm + (size_t)n0 * KK + kk, KK, lane);
        acc[q] = wmma_bf16(a, b, acc[q]);
      }
    }
    int rofs = (lane >> 4) << 3;
#pragma unroll
    for (int q = 0; q < 4; ++q) {
      int n = ((((g4 << 2) + q) << 4)) + (lane & 15);
      float bi = bias[n];
#pragma unroll
      for (int r = 0; r < 8; ++r)
        hid[(m0 + rofs + r) * kNH + n] = f2bf(fmaxf(acc[q][r] + bi, 0.f));
    }
  }
  __syncthreads();
  if (tid < kB * kOUT) {
    int b = tid >> 1, o = tid & 1;
    float s = db[o];
    for (int j = 0; j < kNH; ++j) s += bf2f(hid[b * kNH + j]) * dw[o * kNH + j];
    out[b * kOUT + o] = s;
  }
}

// ---------------- host launch ----------------
extern "C" void kernel_launch(void* const* d_in, const int* in_sizes, int n_in,
                              void* d_out, int out_size, void* d_ws, size_t ws_size,
                              hipStream_t stream) {
  (void)in_sizes; (void)n_in; (void)out_size; (void)ws_size;
  const int*   x     = (const int*)d_in[0];
  const int*   seq   = (const int*)d_in[1];
  const float* emb   = (const float*)d_in[2];
  const float* Wih_f = (const float*)d_in[3];
  const float* Whh_f = (const float*)d_in[4];
  const float* Wih_b = (const float*)d_in[5];
  const float* Whh_b = (const float*)d_in[6];
  const float* S1w   = (const float*)d_in[7];
  const float* S2w   = (const float*)d_in[8];
  const float* MLPw  = (const float*)d_in[9];
  const float* MLPb  = (const float*)d_in[10];
  const float* decw  = (const float*)d_in[11];
  const float* decb  = (const float*)d_in[12];

  char* ws = (char*)d_ws;
  unsigned short* WCF = (unsigned short*)(ws + OFF_WCF);
  unsigned short* WCB = (unsigned short*)(ws + OFF_WCB);
  unsigned short* S1b = (unsigned short*)(ws + OFF_S1);
  unsigned short* S2b = (unsigned short*)(ws + OFF_S2);
  unsigned short* MWb = (unsigned short*)(ws + OFF_MW);
  unsigned short* XF  = (unsigned short*)(ws + OFF_XF);
  unsigned short* XR  = (unsigned short*)(ws + OFF_XR);
  unsigned short* H   = (unsigned short*)(ws + OFF_H);
  float*          LG  = (float*)(ws + OFF_LG);
  unsigned short* BM  = (unsigned short*)(ws + OFF_BM);

  float* out      = (float*)d_out;
  float* outPen   = out + kB * kOUT;        // [128]
  float* outA     = out + kB * kOUT + 1;    // [129..]

  // weight prep
  pack_wcomb_kernel<<<4 * kNH, kKC, 0, stream>>>(Wih_f, Whh_f, WCF);
  pack_wcomb_kernel<<<4 * kNH, kKC, 0, stream>>>(Wih_b, Whh_b, WCB);
  cvt_bf16_kernel<<<(kDA * kH2 + 255) / 256, 256, 0, stream>>>(S1w, S1b, kDA * kH2);
  cvt_bf16_kernel<<<(kR * kDA + 255) / 256, 256, 0, stream>>>(S2w, S2b, kR * kDA);
  cvt_bf16_kernel<<<(kNH * kR * kH2 + 255) / 256, 256, 0, stream>>>(MLPw, MWb, kNH * kR * kH2);

  // embedding gather (fwd + reversed)
  embed_kernel<<<dim3(kS, kB), kEMBP, 0, stream>>>(x, seq, emb, XF, XR);

  // bidirectional LSTM (one persistent workgroup per direction)
  lstm_kernel<<<2, 1024, 0, stream>>>(XF, XR, WCF, WCB, seq, H);

  // attention scores + mask
  attn_kernel<<<(kB * kS) / 64, 256, 0, stream>>>(H, S1b, S2b, seq, LG);

  // softmax -> A (f32, directly into d_out)
  softmax_kernel<<<kB * kR, kS, 0, stream>>>(LG, outA);

  // M = A @ H -> flattened bf16 BM
  bm_kernel<<<kB * kR, kH2, 0, stream>>>(outA, H, BM);

  // penalty
  zero_penal_kernel<<<1, 32, 0, stream>>>(outPen);
  aat_kernel<<<kB, kR * kR, 0, stream>>>(outA, outPen);

  // MLP + decoder -> decoded
  mlp_dec_kernel<<<1, 512, 0, stream>>>(BM, MWb, MLPb, decw, decb, out);
}
